// RelativeMultiHeadSelfAttentionBlock_15564961481306
// MI455X (gfx1250) — matrix operand order repaired and checked by reading, not compiled, not run on hardware
//
#include <hip/hip_runtime.h>
#include <hip/hip_bf16.h>

// ---------------- problem constants ----------------
#define BATCH   2
#define SEQ     4096
#define DMODEL  512
#define HEADS   8
#define DK      64
#define NROWS   (BATCH * SEQ)          // 8192
#define MAXREL  128

// ---------------- vector types ----------------
typedef __attribute__((ext_vector_type(16))) __bf16         v16bf;
typedef __attribute__((ext_vector_type(8)))  float          v8f;
typedef __attribute__((ext_vector_type(8)))  unsigned short v8us;
typedef __attribute__((ext_vector_type(4)))  unsigned int   v4u;
typedef __attribute__((ext_vector_type(8)))  int            v8i;
typedef __attribute__((ext_vector_type(4)))  int            v4i;

struct Pair16 { v8us lo, hi; };          // 32 bytes == v16bf
struct U16x16 { unsigned short u[16]; }; // 32 bytes == v16bf

__device__ __forceinline__ unsigned short f32_to_bf16(float f) {
  unsigned int u = __builtin_bit_cast(unsigned int, f);
  unsigned int r = u + 0x7FFFu + ((u >> 16) & 1u);   // round to nearest even
  return (unsigned short)(r >> 16);
}

// A-fragment (16x32 bf16, row major, leading dim ld):
// lane<16: row=lane, K = {k..k+7, k+16..k+23}; lane>=16: row=lane-16, K = {k+8..k+15, k+24..k+31}
__device__ __forceinline__ v16bf load_a_bf16(const unsigned short* p0, int k, int ld, int lane) {
  const unsigned short* p = p0 + (size_t)(lane & 15) * ld + k + ((lane & 16) ? 8 : 0);
  Pair16 q;
  q.lo = *(const v8us*)p;
  q.hi = *(const v8us*)(p + 16);
  return __builtin_bit_cast(v16bf, q);
}

// B-fragment where memory holds B^T row-major (row = output col n, ld elems):
// lane<16: col=lane, K = {k..k+7, k+16..k+23}; lane>=16: col=lane-16, K = {k+8..k+15, k+24..k+31}
__device__ __forceinline__ v16bf load_bt_bf16(const unsigned short* p0, int k, int ld, int lane) {
  const unsigned short* p = p0 + (size_t)(lane & 15) * ld + k + ((lane & 16) ? 8 : 0);
  Pair16 q;
  q.lo = *(const v8us*)p;
  q.hi = *(const v8us*)(p + 16);
  return __builtin_bit_cast(v16bf, q);
}

// A-fragment from f32 data (attn matrix), non-temporal (stream bypasses L2 retention)
__device__ __forceinline__ v16bf load_a_f32_nt(const float* p0, int k, int ld, int lane) {
  const float* p = p0 + (size_t)(lane & 15) * ld + k + ((lane & 16) ? 8 : 0);
  v8f lo = __builtin_nontemporal_load((const v8f*)p);
  v8f hi = __builtin_nontemporal_load((const v8f*)(p + 16));
  U16x16 t;
#pragma unroll
  for (int i = 0; i < 8; ++i) { t.u[i] = f32_to_bf16(lo[i]); t.u[i + 8] = f32_to_bf16(hi[i]); }
  return __builtin_bit_cast(v16bf, t);
}

#define WMMA_BF16(A, Bf, C) \
  __builtin_amdgcn_wmma_f32_16x16x32_bf16(false, (A), false, (Bf), (short)0, (C), false, false)

// ---------------- Tensor Data Mover: 2D tile load (global -> LDS) ----------------
// D# per CDNA5 ISA: group0 {count,lds_addr,global_addr,type=2},
// group1 {data_size=2B, tensor_dim0/1, tile_dim0/1, tensor_dim0_stride}. 2D => groups 2/3 zero.
// This toolchain exposes the 6-arg builtin: (v4u, v8i, v4i, v4i, v8i, i32 cpol).
__device__ __forceinline__ void tdm_load_2d(unsigned lds_off, const void* gaddr,
                                            unsigned tdim0, unsigned tdim1,
                                            unsigned stride0_elems,
                                            unsigned tile0, unsigned tile1) {
  unsigned long long ga = (unsigned long long)gaddr;
  v4u g0;
  g0[0] = 1u;                                            // count=1, is_restore=0
  g0[1] = lds_off;                                       // lds_addr (bytes)
  g0[2] = (unsigned)(ga & 0xFFFFFFFFu);                  // global_addr[31:0]
  g0[3] = (unsigned)((ga >> 32) & 0x01FFFFFFu)           // global_addr[56:32]
        | (2u << 30);                                    // type=2 (image)
  v8i g1;
  g1[0] = (int)(1u << 16);                               // data_size=1 => 2 bytes; wg_mask=0
  g1[1] = (int)((tdim0 & 0xFFFFu) << 16);                // tensor_dim0[15:0] (atomic_barrier=0)
  g1[2] = (int)(((tdim0 >> 16) & 0xFFFFu) | ((tdim1 & 0xFFFFu) << 16));
  g1[3] = (int)(((tdim1 >> 16) & 0xFFFFu) | ((tile0 & 0xFFFFu) << 16)); // tile_dim0
  g1[4] = (int)(tile1 & 0xFFFFu);                        // tile_dim1 (tile_dim2=0)
  g1[5] = (int)stride0_elems;                            // tensor_dim0_stride[31:0]
  g1[6] = 0;                                             // stride0[47:32] | stride1[15:0]
  g1[7] = 0;
  v4i g2 = {0, 0, 0, 0};
  v4i g3 = {0, 0, 0, 0};
  v8i g4 = {0, 0, 0, 0, 0, 0, 0, 0};
  __builtin_amdgcn_tensor_load_to_lds(g0, g1, g2, g3, g4, 0);
}

__device__ __forceinline__ unsigned lds_offset_of(const void* p) {
  return (unsigned)(unsigned long long)p;                // generic ptr low 32 bits = LDS offset
}

// ---------------- kernel 0: f32 -> bf16 weight convert ----------------
__global__ void k_cvt(const float* __restrict__ x, unsigned short* __restrict__ y, int n) {
  int i = blockIdx.x * blockDim.x + threadIdx.x;
  if (i < n) y[i] = f32_to_bf16(x[i]);
}

// ---------------- kernel 1: LayerNorm, one wave per 512-wide row ----------------
__global__ void k_layernorm(const float* __restrict__ x, const float* __restrict__ g,
                            const float* __restrict__ be, unsigned short* __restrict__ y) {
  int wid  = (blockIdx.x * blockDim.x + threadIdx.x) >> 5;   // row id
  int lane = threadIdx.x & 31;
  const float* row = x + (size_t)wid * DMODEL;
  float v[16], s = 0.f, ss = 0.f;
#pragma unroll
  for (int i = 0; i < 16; ++i) {
    float t = row[lane + i * 32];
    v[i] = t; s += t; ss += t * t;
  }
#pragma unroll
  for (int m = 16; m >= 1; m >>= 1) { s += __shfl_xor(s, m, 32); ss += __shfl_xor(ss, m, 32); }
  float mu  = s * (1.f / DMODEL);
  float var = ss * (1.f / DMODEL) - mu * mu;
  float inv = rsqrtf(var + 1e-5f);
  unsigned short* yr = y + (size_t)wid * DMODEL;
#pragma unroll
  for (int i = 0; i < 16; ++i) {
    int c = lane + i * 32;
    yr[c] = f32_to_bf16((v[i] - mu) * inv * g[c] + be[c]);
  }
}

// ---------------- kernel 2: projection GEMM Y = Xn @ W^T + b  (bf16 out) ----------------
// block = one 16-row M tile staged to LDS by TDM; 8 waves cover all 512 N (64 each)
__global__ void k_proj(const unsigned short* __restrict__ A,  // [8192,512] bf16
                       const unsigned short* __restrict__ W,  // [512,512]  bf16 row-major
                       const float* __restrict__ bias,        // [512]
                       unsigned short* __restrict__ Y, int transposed) {
  __shared__ alignas(16) unsigned short shA[16 * DMODEL];     // 16 KB
  int m0   = blockIdx.x << 4;                                 // uniform (SGPR) for TDM D#
  int lane = threadIdx.x & 31;
  int ng   = threadIdx.x >> 5;                                // wave id: N-group of 64
  if (threadIdx.x < 32) {
    tdm_load_2d(lds_offset_of(shA), A + (size_t)m0 * DMODEL,
                DMODEL, 16, DMODEL, DMODEL, 16);
    __builtin_amdgcn_s_wait_tensorcnt(0);
  }
  __syncthreads();
  v8f c[4] = {};
  for (int k = 0; k < DMODEL; k += 32) {
    v16bf a = load_a_bf16(shA, k, DMODEL, lane);              // ds_load_b128 path
#pragma unroll
    for (int t = 0; t < 4; ++t) {
      v16bf bf = load_bt_bf16(W + (size_t)((ng * 4 + t) * 16) * DMODEL, k, DMODEL, lane);
      c[t] = WMMA_BF16(a, bf, c[t]);
    }
  }
  int mb = m0 + ((lane & 16) ? 8 : 0);
#pragma unroll
  for (int t = 0; t < 4; ++t) {
    int n = (ng * 4 + t) * 16 + (lane & 15);
    float bn = bias[n];
    int h = n >> 6, d = n & 63;
#pragma unroll
    for (int r = 0; r < 8; ++r) {
      int m = mb + r;
      int b = m >> 12, s = m & (SEQ - 1);
      size_t idx = transposed
        ? ((((size_t)b * HEADS + h) * DK + d) * SEQ + s)
        : ((((size_t)b * HEADS + h) * SEQ + s) * DK + d);
      Y[idx] = f32_to_bf16(c[t][r] + bn);
    }
  }
}

// ---------------- kernel 3: scores = QK^T/8 + relbias, mask; raw into attn buffer ----------------
__global__ void k_scores(const unsigned short* __restrict__ Qh,  // [B,H,S,DK]
                         const unsigned short* __restrict__ Kh,  // [B,H,S,DK]
                         const float* __restrict__ rel,          // [255,8]
                         const unsigned char* __restrict__ mask, // [B,S]
                         float* __restrict__ Sc) {               // [B,H,S,S]
  __shared__ alignas(16) unsigned short shQ[16 * DK];            // 2 KB shared Q tile
  int lane = threadIdx.x & 31;
  int wv   = threadIdx.x >> 5;              // 8 waves: consecutive key tiles
  int blk  = blockIdx.x;
  int ngrp = blk & 31;                      // 32 groups * 8 waves = 256 key tiles
  int qt   = (blk >> 5) & 255;
  int bh   = blk >> 13;                     // 0..15
  int h = bh & 7, b = bh >> 3;
  int n0 = (ngrp * 8 + wv) * 16;
  int m0 = qt * 16;
  const unsigned short* Qb = Qh + (size_t)bh * SEQ * DK;
  const unsigned short* Kb = Kh + (size_t)bh * SEQ * DK;
  if (threadIdx.x < 32) {
    tdm_load_2d(lds_offset_of(shQ), Qb + (size_t)m0 * DK, DK, 16, DK, DK, 16);
    __builtin_amdgcn_s_wait_tensorcnt(0);
  }
  __syncthreads();
  v8f c = {};
#pragma unroll
  for (int k = 0; k < DK; k += 32) {
    v16bf a  = load_a_bf16(shQ, k, DK, lane);
    v16bf bf = load_bt_bf16(Kb + (size_t)n0 * DK, k, DK, lane);
    c = WMMA_BF16(a, bf, c);
  }
  int j = n0 + (lane & 15);
  int masked = mask[b * SEQ + j];
  int mb = m0 + ((lane & 16) ? 8 : 0);
  float* out = Sc + ((size_t)bh * SEQ + mb) * SEQ + j;
#pragma unroll
  for (int r = 0; r < 8; ++r) {
    int dr = (mb + r) - j;
    dr = dr > (MAXREL - 1) ? (MAXREL - 1) : (dr < -(MAXREL - 1) ? -(MAXREL - 1) : dr);
    float val = c[r] * 0.125f + rel[(dr + MAXREL - 1) * HEADS + h];
    if (masked) val = -__builtin_inff();
    __builtin_nontemporal_store(val, &out[(size_t)r * SEQ]);   // 1GB stream: keep out of L2
  }
}

// ---------------- kernel 4: in-place row softmax, one 256-thread block per 4096 row ----------------
__global__ void k_softmax(float* __restrict__ Sc) {
  __shared__ float red[8];
  int t = threadIdx.x;
  float* row = Sc + (size_t)blockIdx.x * SEQ;
  float v[16];
  float mx = -__builtin_inff();
#pragma unroll
  for (int i = 0; i < 16; ++i) {
    v[i] = __builtin_nontemporal_load(&row[t + i * 256]);
    mx = fmaxf(mx, v[i]);
  }
#pragma unroll
  for (int m = 16; m >= 1; m >>= 1) mx = fmaxf(mx, __shfl_xor(mx, m, 32));
  if ((t & 31) == 0) red[t >> 5] = mx;
  __syncthreads();
  float m2 = red[0];
#pragma unroll
  for (int i = 1; i < 8; ++i) m2 = fmaxf(m2, red[i]);
  __syncthreads();
  float s = 0.f;
#pragma unroll
  for (int i = 0; i < 16; ++i) { v[i] = __expf(v[i] - m2); s += v[i]; }
#pragma unroll
  for (int m = 16; m >= 1; m >>= 1) s += __shfl_xor(s, m, 32);
  if ((t & 31) == 0) red[t >> 5] = s;
  __syncthreads();
  float tot = 0.f;
#pragma unroll
  for (int i = 0; i < 8; ++i) tot += red[i];
  float inv = 1.f / tot;
#pragma unroll
  for (int i = 0; i < 16; ++i)
    __builtin_nontemporal_store(v[i] * inv, &row[t + i * 256]);
}

// ---------------- kernel 5: head_out = attn @ V  (wave does 16x64, attn read once) ----------------
__global__ void k_av(const float* __restrict__ Attn,          // [B,H,S,S] softmaxed
                     const unsigned short* __restrict__ Vt,   // [B,H,DK,S] (transposed)
                     unsigned short* __restrict__ Hd) {       // [B,S,512] bf16
  int lane = threadIdx.x & 31;
  int wid  = (blockIdx.x * blockDim.x + threadIdx.x) >> 5;
  int qt = wid & 255;
  int bh = wid >> 8;                       // 0..15
  int h = bh & 7, b = bh >> 3;
  const float* Ab = Attn + ((size_t)bh * SEQ + qt * 16) * SEQ;
  const unsigned short* Vb = Vt + (size_t)bh * DK * SEQ;
  v8f c[4] = {};
  for (int k = 0; k < SEQ; k += 32) {
    __builtin_prefetch(Ab + (size_t)(lane & 15) * SEQ + k + 128, 0, 0);
    v16bf a = load_a_f32_nt(Ab, k, SEQ, lane);
#pragma unroll
    for (int t = 0; t < 4; ++t) {
      v16bf bf = load_bt_bf16(Vb + (size_t)(t * 16) * SEQ, k, SEQ, lane);
      c[t] = WMMA_BF16(a, bf, c[t]);
    }
  }
  int mb = qt * 16 + ((lane & 16) ? 8 : 0);
#pragma unroll
  for (int t = 0; t < 4; ++t) {
    int d = t * 16 + (lane & 15);
#pragma unroll
    for (int r = 0; r < 8; ++r)
      Hd[((size_t)b * SEQ + (mb + r)) * DMODEL + h * DK + d] = f32_to_bf16(c[t][r]);
  }
}

// ---------------- kernel 6: out = Hd @ wo^T + bo (f32 out), TDM-staged A tile ----------------
__global__ void k_out(const unsigned short* __restrict__ A,   // [8192,512] bf16
                      const unsigned short* __restrict__ W,   // [512,512] bf16
                      const float* __restrict__ bias,
                      float* __restrict__ Y) {                // [8192,512] f32
  __shared__ alignas(16) unsigned short shA[16 * DMODEL];
  int m0   = blockIdx.x << 4;
  int lane = threadIdx.x & 31;
  int ng   = threadIdx.x >> 5;
  if (threadIdx.x < 32) {
    tdm_load_2d(lds_offset_of(shA), A + (size_t)m0 * DMODEL,
                DMODEL, 16, DMODEL, DMODEL, 16);
    __builtin_amdgcn_s_wait_tensorcnt(0);
  }
  __syncthreads();
  v8f c[4] = {};
  for (int k = 0; k < DMODEL; k += 32) {
    v16bf a = load_a_bf16(shA, k, DMODEL, lane);
#pragma unroll
    for (int t = 0; t < 4; ++t) {
      v16bf bf = load_bt_bf16(W + (size_t)((ng * 4 + t) * 16) * DMODEL, k, DMODEL, lane);
      c[t] = WMMA_BF16(a, bf, c[t]);
    }
  }
  int mb = m0 + ((lane & 16) ? 8 : 0);
#pragma unroll
  for (int t = 0; t < 4; ++t) {
    int n = (ng * 4 + t) * 16 + (lane & 15);
    float bn = bias[n];
#pragma unroll
    for (int r = 0; r < 8; ++r)
      Y[(size_t)(mb + r) * DMODEL + n] = c[t][r] + bn;
  }
}

// ---------------- host side ----------------
extern "C" void kernel_launch(void* const* d_in, const int* in_sizes, int n_in,
                              void* d_out, int out_size, void* d_ws, size_t ws_size,
                              hipStream_t stream) {
  (void)in_sizes; (void)n_in; (void)out_size; (void)ws_size;
  const float* q   = (const float*)d_in[0];
  const float* k   = (const float*)d_in[1];
  const float* v   = (const float*)d_in[2];
  const unsigned char* kpm = (const unsigned char*)d_in[3];
  const float* g   = (const float*)d_in[4];
  const float* be  = (const float*)d_in[5];
  const float* wq  = (const float*)d_in[6];  const float* bq = (const float*)d_in[7];
  const float* wk  = (const float*)d_in[8];  const float* bk = (const float*)d_in[9];
  const float* wv  = (const float*)d_in[10]; const float* bv = (const float*)d_in[11];
  const float* wo  = (const float*)d_in[12]; const float* bo = (const float*)d_in[13];
  const float* rel = (const float*)d_in[14];

  float* out  = (float*)d_out;                      // [B,S,512]
  float* attn = out + (size_t)NROWS * DMODEL;       // [B,H,S,S]

  // workspace partition (bf16 stored as ushort)
  unsigned short* lnq = (unsigned short*)d_ws;
  unsigned short* lnk = lnq + (size_t)NROWS * DMODEL;
  unsigned short* lnv = lnk + (size_t)NROWS * DMODEL;
  unsigned short* wqb = lnv + (size_t)NROWS * DMODEL;
  unsigned short* wkb = wqb + (size_t)DMODEL * DMODEL;
  unsigned short* wvb = wkb + (size_t)DMODEL * DMODEL;
  unsigned short* wob = wvb + (size_t)DMODEL * DMODEL;
  unsigned short* Qh  = wob + (size_t)DMODEL * DMODEL;      // [B,H,S,DK]
  unsigned short* Kh  = Qh  + (size_t)NROWS * DMODEL;
  unsigned short* Vt  = Kh  + (size_t)NROWS * DMODEL;       // [B,H,DK,S]
  unsigned short* Hd  = Vt  + (size_t)NROWS * DMODEL;       // [B,S,512]

  const int nw = DMODEL * DMODEL;
  k_cvt<<<nw / 256, 256, 0, stream>>>(wq, wqb, nw);
  k_cvt<<<nw / 256, 256, 0, stream>>>(wk, wkb, nw);
  k_cvt<<<nw / 256, 256, 0, stream>>>(wv, wvb, nw);
  k_cvt<<<nw / 256, 256, 0, stream>>>(wo, wob, nw);

  k_layernorm<<<NROWS / 8, 256, 0, stream>>>(q, g, be, lnq);
  k_layernorm<<<NROWS / 8, 256, 0, stream>>>(k, g, be, lnk);
  k_layernorm<<<NROWS / 8, 256, 0, stream>>>(v, g, be, lnv);

  // one block per 16-row M tile (TDM-staged), 8 waves cover N=512
  k_proj<<<512, 256, 0, stream>>>(lnq, wqb, bq, Qh, 0);
  k_proj<<<512, 256, 0, stream>>>(lnk, wkb, bk, Kh, 0);
  k_proj<<<512, 256, 0, stream>>>(lnv, wvb, bv, Vt, 1);

  // 16 bh * 256 qtiles * 32 ngroups = 131072 blocks
  k_scores<<<BATCH * HEADS * 256 * 32, 256, 0, stream>>>(Qh, Kh, rel, kpm, attn);

  // one block per attention row
  k_softmax<<<BATCH * HEADS * SEQ, 256, 0, stream>>>(attn);

  // 16 bh * 256 qtiles = 4096 waves = 512 blocks
  k_av<<<512, 256, 0, stream>>>(attn, Vt, Hd);

  k_out<<<512, 256, 0, stream>>>(Hd, wob, bo, out);
}
